// Conv2d_mvm_82016695484430
// MI455X (gfx1250) — compile-verified, matrix operand order checked by reading, not run on hardware
//
#include <hip/hip_runtime.h>
#include <stdint.h>

// ---------------------------------------------------------------------------
// PUMA crossbar conv collapses exactly to:
//   x16[p,l]  = sext16(rint(x_patch * 4096))           (two's complement wrap)
//   weff[c,l] = sign(w) * min(rint(|w|*4096), 65535)
//   acc[p,c]  = sum_l x16*weff      (int32)
//   out       = clamp(round_half_even(acc/4096), -32768, 32767) / 4096
// M = 4*16*16 = 1024, N = 128, K = 64*9 = 576 = 9 * 64.
// int16 x int16 done as 4x IU8 WMMA (hi/lo byte decomposition).
// ---------------------------------------------------------------------------

typedef __attribute__((ext_vector_type(8))) int v8i;

#define M_TILES    64          // 1024 / 16
#define N_TILES    8           // 128 / 16
#define KSTEPS     9           // 576 / 64
#define FRAG_BYTES 1024        // 32 lanes * 32 bytes (v8i per lane)

static constexpr size_t A_LO_OFF = 0;
static constexpr size_t A_HI_OFF = (size_t)M_TILES * KSTEPS * FRAG_BYTES;             //  589824
static constexpr size_t B_LO_OFF = 2 * A_HI_OFF;                                      // 1179648
static constexpr size_t B_HI_OFF = B_LO_OFF + (size_t)N_TILES * KSTEPS * FRAG_BYTES;  // 1253376
// total ws use: 1327104 bytes

// 8-bit A-matrix 16x64 VGPR byte map (ISA 7.12.2), mirrored for B:
//   k(vgpr,j,half) = 32*(v>>2) + 16*((v>>1)&1) + 8*half + 4*(v&1) + j
// Inverse, given kk in [0,64):
__device__ __forceinline__ void kk_to_vhj(int kk, int& vr, int& half, int& j) {
    vr   = ((kk >> 5) & 1) * 4 + ((kk >> 4) & 1) * 2 + ((kk >> 2) & 1);
    half = (kk >> 3) & 1;
    j    = kk & 3;
}

// ---------------- weight pack: w[128,64,3,3] -> B fragments -----------------
__global__ void pack_w_kernel(const float* __restrict__ w,
                              uint8_t* __restrict__ blo,
                              uint8_t* __restrict__ bhi) {
    int idx = blockIdx.x * blockDim.x + threadIdx.x;   // c*576 + l, l=(cin,ki,kj)
    if (idx >= 128 * 576) return;
    int c = idx / 576, l = idx - c * 576;

    float wv  = w[idx];
    int   mi  = (int)rintf(fabsf(wv) * 4096.0f);       // RNE, matches jnp.round
    mi        = mi > 65535 ? 65535 : mi;
    int  weff = (wv < 0.0f) ? -mi : mi;

    int ks = l >> 6, kk = l & 63;
    int vr, half, j;
    kk_to_vhj(kk, vr, half, j);
    int lane = half * 16 + (c & 15);                   // N = lane&15
    int n0   = c >> 4;
    size_t off = (((size_t)(ks * N_TILES + n0) * 32 + lane) * 32) + vr * 4 + j;
    blo[off] = (uint8_t)(weff & 0xFF);
    bhi[off] = (uint8_t)((weff >> 8) & 0xFF);
}

// ------------- im2col + quantize: x[4,64,16,16] -> A fragments --------------
__global__ void pack_x_kernel(const float* __restrict__ x,
                              uint8_t* __restrict__ alo,
                              uint8_t* __restrict__ ahi) {
    int idx = blockIdx.x * blockDim.x + threadIdx.x;   // p*576 + l
    if (idx >= 1024 * 576) return;
    int p = idx / 576, l = idx - p * 576;

    int b = p >> 8, oh = (p >> 4) & 15, ow = p & 15;
    int cin = l / 9, rem = l - cin * 9, ki = rem / 3, kj = rem - ki * 3;
    int ih = oh - 1 + ki, iw = ow - 1 + kj;            // pad = 1

    float xv = 0.0f;
    if ((unsigned)ih < 16u && (unsigned)iw < 16u)
        xv = x[(((size_t)b * 64 + cin) * 16 + ih) * 16 + iw];

    int xi  = (int)rintf(xv * 4096.0f);
    int x16 = (int)(short)(xi & 0xFFFF);               // two's complement wrap

    int ks = l >> 6, kk = l & 63;
    int vr, half, j;
    kk_to_vhj(kk, vr, half, j);
    int lane = half * 16 + (p & 15);                   // M = lane&15
    int m0   = p >> 4;
    size_t off = (((size_t)(m0 * KSTEPS + ks) * 32 + lane) * 32) + vr * 4 + j;
    alo[off] = (uint8_t)(x16 & 0xFF);
    ahi[off] = (uint8_t)((x16 >> 8) & 0xFF);
}

// ----------------------------- IU8 WMMA GEMM --------------------------------
__global__ void __launch_bounds__(128)
gemm_iu8_kernel(const uint8_t* __restrict__ ws, float* __restrict__ out) {
    const int lane = threadIdx.x & 31;
    const int wave = threadIdx.x >> 5;
    const int tile = blockIdx.x * 4 + wave;            // 512 tiles, 4 waves/block
    const int m0   = tile >> 3;                        // 0..63
    const int n0   = tile & 7;                         // 0..7

    const uint8_t* alo = ws + A_LO_OFF;
    const uint8_t* ahi = ws + A_HI_OFF;
    const uint8_t* blo = ws + B_LO_OFF;
    const uint8_t* bhi = ws + B_HI_OFF;

    v8i hh = {}, hl = {}, lh = {}, ll = {};

#pragma unroll
    for (int ks = 0; ks < KSTEPS; ++ks) {
        const size_t aoff = ((size_t)(m0 * KSTEPS + ks) * 32 + lane) * 32;
        const size_t boff = ((size_t)(ks * N_TILES + n0) * 32 + lane) * 32;
        const v8i aL = *(const v8i*)(alo + aoff);
        const v8i aH = *(const v8i*)(ahi + aoff);
        const v8i bL = *(const v8i*)(blo + boff);
        const v8i bH = *(const v8i*)(bhi + boff);
        if (ks < KSTEPS - 1) {                          // uniform: exercises global_prefetch_b8
            __builtin_prefetch(alo + aoff + FRAG_BYTES, 0, 1);
            __builtin_prefetch(ahi + aoff + FRAG_BYTES, 0, 1);
        }
        // (sgn_a, A, sgn_b, B, C, reuse_a, reuse_b)
        hh = __builtin_amdgcn_wmma_i32_16x16x64_iu8(true,  aH, true,  bH, hh, false, false);
        hl = __builtin_amdgcn_wmma_i32_16x16x64_iu8(true,  aH, false, bL, hl, false, false);
        lh = __builtin_amdgcn_wmma_i32_16x16x64_iu8(false, aL, true,  bH, lh, false, false);
        ll = __builtin_amdgcn_wmma_i32_16x16x64_iu8(false, aL, false, bL, ll, false, false);
    }

    // C/D 16x16 i32 layout: VGPR i -> M = i + 8*(lane>=16), N = lane&15
    const int nloc = lane & 15;
    const int half = lane >> 4;
#pragma unroll
    for (int i = 0; i < 8; ++i) {
        long long acc = ((long long)hh[i] << 16)
                      + ((long long)(hl[i] + lh[i]) << 8)
                      + (long long)ll[i];
        // round-half-even of acc/4096, then clamp to int16 (ACM stage)
        long long q = acc >> 12;                        // floor
        int rrem = (int)(acc & 4095);
        if (rrem > 2048 || (rrem == 2048 && (q & 1))) q += 1;
        if (q < -32768) q = -32768;
        if (q >  32767) q =  32767;

        int p = m0 * 16 + i + half * 8;                 // patch index
        int c = n0 * 16 + nloc;                         // output channel
        int b = p >> 8, r = p & 255;                    // r = oh*16+ow
        out[((size_t)b * 128 + c) * 256 + r] = (float)q * (1.0f / 4096.0f);
    }
}

// ---------------------------------------------------------------------------
extern "C" void kernel_launch(void* const* d_in, const int* in_sizes, int n_in,
                              void* d_out, int out_size, void* d_ws, size_t ws_size,
                              hipStream_t stream) {
    const float* x = (const float*)d_in[0];   // [4,64,16,16]
    const float* w = (const float*)d_in[1];   // [128,64,3,3]
    uint8_t* ws = (uint8_t*)d_ws;
    float* out = (float*)d_out;               // [4,128,16,16]

    pack_w_kernel<<<(128 * 576 + 255) / 256, 256, 0, stream>>>(
        w, ws + B_LO_OFF, ws + B_HI_OFF);
    pack_x_kernel<<<(1024 * 576 + 255) / 256, 256, 0, stream>>>(
        x, ws + A_LO_OFF, ws + A_HI_OFF);
    gemm_iu8_kernel<<<M_TILES * N_TILES / 4, 128, 0, stream>>>(ws, out);
}
